// Forward_64132451664296
// MI455X (gfx1250) — matrix operand order repaired and checked by reading, not compiled, not run on hardware
//
#include <hip/hip_runtime.h>

// ---------------------------------------------------------------------------
// Fused: h = relu(W2·(sigmoid(W1·rmsnorm(W0·x+b0)+b1)+rmsnorm(...)) + b2) + ...
// bf16 WMMA (v_wmma_f32_16x16x32_bf16), f32 accumulate, single pass over x.
// ---------------------------------------------------------------------------

typedef __attribute__((ext_vector_type(16))) __bf16          v16bf;
typedef __attribute__((ext_vector_type(8)))  float           v8f;
typedef __attribute__((ext_vector_type(8)))  unsigned short  u16x8;
typedef __attribute__((ext_vector_type(16))) unsigned short  u16x16;
typedef __attribute__((ext_vector_type(4)))  unsigned short  u16x4;

union Frag {
  u16x16 u;
  v16bf  b;
  u16x8  h[2];
};

__device__ __forceinline__ unsigned short f32_to_bf16(float f) {
  unsigned int u = __float_as_uint(f);
  u += 0x7FFFu + ((u >> 16) & 1u);   // round-to-nearest-even
  return (unsigned short)(u >> 16);
}

#define DIM_D 1024
#define DIM_H 128
#define ASTR  1032   // 1024 + 8 ushort pad -> row stride 2064B, conflict-free b128
#define HSTR  136    // 128 + 8 ushort pad  -> row stride 272B

// ---- weight prep: f32 [K][N] -> bf16 transposed [N][K] ---------------------
__global__ void wt_bf16_kernel(const float* __restrict__ W,
                               unsigned short* __restrict__ WT,
                               int K, int N) {
  int idx = blockIdx.x * blockDim.x + threadIdx.x;
  if (idx >= K * N) return;
  int k = idx / N;
  int n = idx - k * N;
  WT[(size_t)n * K + k] = f32_to_bf16(W[idx]);
}

// ---- main fused kernel: one block = 16 rows, one wave = one 16-col tile ----
__global__ __launch_bounds__(256) void fused_mlp_kernel(
    const float* __restrict__ x,
    const unsigned short* __restrict__ W0T, const float* __restrict__ b0,
    const unsigned short* __restrict__ W1T, const float* __restrict__ b1,
    const unsigned short* __restrict__ W2T, const float* __restrict__ b2,
    float* __restrict__ out) {
  __shared__ unsigned short Abf[16 * ASTR];   // x tile, bf16   (~33 KB)
  __shared__ unsigned short Hbf[16 * HSTR];   // activation tile, bf16
  __shared__ float sumsq[16];
  __shared__ float scl[16];

  const int tid  = threadIdx.x;
  const int lane = tid & 31;
  const int wv   = tid >> 5;          // 8 waves
  const int n0   = wv * 16;           // wave's N-tile base
  const int nn   = lane & 15;         // M (for A frag) / N (for B,C frags)
  const int sel  = lane >> 4;         // half-wave selector
  const size_t row0 = (size_t)blockIdx.x * 16;

  // ---- stage x tile (16 x 1024 f32) into LDS as bf16 ----
  {
    const int r  = tid >> 4;
    const int c0 = (tid & 15) * 4;
    const float* xrow = x + (row0 + r) * DIM_D;
    unsigned short* arow = Abf + r * ASTR;
#pragma unroll
    for (int it = 0; it < 16; ++it) {
      const int c = c0 + it * 64;
      const float4 v = *(const float4*)(xrow + c);
      u16x4 p;
      p.x = f32_to_bf16(v.x);
      p.y = f32_to_bf16(v.y);
      p.z = f32_to_bf16(v.z);
      p.w = f32_to_bf16(v.w);
      *(u16x4*)(arow + c) = p;
    }
  }
  if (tid < 16) sumsq[tid] = 0.0f;
  __syncthreads();

  // ---- GEMM1: h0 = x @ W0 + b0  (K = 1024, 32 WMMA steps) ----
  v8f c1 = {0.f, 0.f, 0.f, 0.f, 0.f, 0.f, 0.f, 0.f};
#pragma unroll 4
  for (int kk = 0; kk < DIM_D; kk += 32) {
    Frag a, b;
    const unsigned short* ap = Abf + nn * ASTR + kk + sel * 8;
    a.h[0] = *(const u16x8*)(ap);          // K = kk + 8*sel .. +7
    a.h[1] = *(const u16x8*)(ap + 16);     // K = kk + 16 + 8*sel .. +7
    const unsigned short* bp = W0T + (size_t)(n0 + nn) * DIM_D + kk + sel * 8;
    b.h[0] = *(const u16x8*)(bp);          // K = kk + 8*sel .. +7
    b.h[1] = *(const u16x8*)(bp + 16);     // K = kk + 16 + 8*sel .. +7
    c1 = __builtin_amdgcn_wmma_f32_16x16x32_bf16(false, a.b, false, b.b,
                                                 (short)0, c1, false, false);
  }
  const float bias0 = b0[n0 + nn];
#pragma unroll
  for (int r = 0; r < 8; ++r) c1[r] += bias0;

  // ---- RMSNorm: sum of squares per row, across 16 lanes + 8 waves ----
#pragma unroll
  for (int r = 0; r < 8; ++r) {
    float s = c1[r] * c1[r];
    s += __shfl_xor(s, 1, 32);
    s += __shfl_xor(s, 2, 32);
    s += __shfl_xor(s, 4, 32);
    s += __shfl_xor(s, 8, 32);
    if (nn == 0) atomicAdd(&sumsq[r + sel * 8], s);   // ds_add_f32
  }
  __syncthreads();
  if (tid < 16) scl[tid] = rsqrtf(sumsq[tid] * (1.0f / 128.0f) + 1e-6f);
  __syncthreads();

  float h0n[8];
#pragma unroll
  for (int r = 0; r < 8; ++r) {
    const int m = r + sel * 8;
    h0n[r] = c1[r] * scl[m];
    Hbf[m * HSTR + n0 + nn] = f32_to_bf16(h0n[r]);
  }
  __syncthreads();

  // ---- GEMM2: z = h0n @ W1 + b1 ; h1 = sigmoid(z) + h0n ----
  v8f c2 = {0.f, 0.f, 0.f, 0.f, 0.f, 0.f, 0.f, 0.f};
#pragma unroll
  for (int kk = 0; kk < DIM_H; kk += 32) {
    Frag a, b;
    const unsigned short* ap = Hbf + nn * HSTR + kk + sel * 8;
    a.h[0] = *(const u16x8*)(ap);
    a.h[1] = *(const u16x8*)(ap + 16);
    const unsigned short* bp = W1T + (n0 + nn) * DIM_H + kk + sel * 8;
    b.h[0] = *(const u16x8*)(bp);
    b.h[1] = *(const u16x8*)(bp + 16);
    c2 = __builtin_amdgcn_wmma_f32_16x16x32_bf16(false, a.b, false, b.b,
                                                 (short)0, c2, false, false);
  }
  const float bias1 = b1[n0 + nn];
  float h1[8];
#pragma unroll
  for (int r = 0; r < 8; ++r) {
    const float z = c2[r] + bias1;
    h1[r] = 1.0f / (1.0f + __expf(-z)) + h0n[r];
  }
  __syncthreads();   // all waves finished reading Hbf before overwrite
#pragma unroll
  for (int r = 0; r < 8; ++r)
    Hbf[(r + sel * 8) * HSTR + n0 + nn] = f32_to_bf16(h1[r]);
  __syncthreads();

  // ---- GEMM3: z = h1 @ W2 + b2 ; out = relu(z) + h1 ----
  v8f c3 = {0.f, 0.f, 0.f, 0.f, 0.f, 0.f, 0.f, 0.f};
#pragma unroll
  for (int kk = 0; kk < DIM_H; kk += 32) {
    Frag a, b;
    const unsigned short* ap = Hbf + nn * HSTR + kk + sel * 8;
    a.h[0] = *(const u16x8*)(ap);
    a.h[1] = *(const u16x8*)(ap + 16);
    const unsigned short* bp = W2T + (n0 + nn) * DIM_H + kk + sel * 8;
    b.h[0] = *(const u16x8*)(bp);
    b.h[1] = *(const u16x8*)(bp + 16);
    c3 = __builtin_amdgcn_wmma_f32_16x16x32_bf16(false, a.b, false, b.b,
                                                 (short)0, c3, false, false);
  }
  const float bias2 = b2[n0 + nn];
  float* orow = out + row0 * DIM_H;
#pragma unroll
  for (int r = 0; r < 8; ++r) {
    const int m = r + sel * 8;
    const float z = c3[r] + bias2;
    orow[(size_t)m * DIM_H + n0 + nn] = fmaxf(z, 0.0f) + h1[r];
  }
}

// ---------------------------------------------------------------------------
extern "C" void kernel_launch(void* const* d_in, const int* in_sizes, int n_in,
                              void* d_out, int out_size, void* d_ws, size_t ws_size,
                              hipStream_t stream) {
  const float* x  = (const float*)d_in[0];
  const float* W0 = (const float*)d_in[1];
  const float* b0 = (const float*)d_in[2];
  const float* W1 = (const float*)d_in[3];
  const float* b1 = (const float*)d_in[4];
  const float* W2 = (const float*)d_in[5];
  const float* b2 = (const float*)d_in[6];
  float* out = (float*)d_out;

  // workspace: bf16 transposed weights (320 KB total)
  unsigned short* W0T = (unsigned short*)d_ws;           // [128][1024]
  unsigned short* W1T = W0T + 1024 * 128;                // [128][128]
  unsigned short* W2T = W1T + 128 * 128;                 // [128][128]

  wt_bf16_kernel<<<(1024 * 128 + 255) / 256, 256, 0, stream>>>(W0, W0T, 1024, 128);
  wt_bf16_kernel<<<(128 * 128 + 255) / 256, 256, 0, stream>>>(W1, W1T, 128, 128);
  wt_bf16_kernel<<<(128 * 128 + 255) / 256, 256, 0, stream>>>(W2, W2T, 128, 128);

  const int rows = in_sizes[0] / DIM_D;   // B*S = 32768
  fused_mlp_kernel<<<rows / 16, 256, 0, stream>>>(x, W0T, b0, W1T, b1, W2T, b2, out);
}
